// MultiHeadAttention_88450556494578
// MI455X (gfx1250) — matrix operand order repaired and checked by reading, not compiled, not run on hardware
//
#include <hip/hip_runtime.h>

// Problem constants (match reference)
#define BQ 8
#define TN 128
#define CD 2048
#define HH 16
#define TP 2048
#define DD 128
#define TT (TP + TN)   // 2176
#define QSCALE 0.08838834764831845f  // 1/sqrt(128)

typedef __bf16 bf16;
typedef __bf16 v16bf __attribute__((ext_vector_type(16)));
typedef __bf16 v8bf  __attribute__((ext_vector_type(8)));
typedef float  v8f   __attribute__((ext_vector_type(8)));

static __device__ __forceinline__ v8f wmma_bf16(v16bf a, v16bf b, v8f c) {
  // D = A(16x32 bf16) * B(32x16 bf16) + C(16x16 f32)
  return __builtin_amdgcn_wmma_f32_16x16x32_bf16(
      /*neg_a=*/false, a, /*neg_b=*/false, b,
      /*c_mod=*/(short)0, c, /*reuse_a=*/false, /*reuse_b=*/false);
}

static __device__ __forceinline__ v8f zero8() {
  v8f r;
#pragma unroll
  for (int e = 0; e < 8; ++e) r[e] = 0.0f;
  return r;
}

static __device__ __forceinline__ v16bf cat16(v8bf lo, v8bf hi) {
  return __builtin_shufflevector(lo, hi, 0, 1, 2, 3, 4, 5, 6, 7,
                                 8, 9, 10, 11, 12, 13, 14, 15);
}

// ---------------------------------------------------------------------------
// Stream K_past/V_past (B,H,TP,D) into the TT-strided cache (B,H,TT,D) using
// the gfx1250 async global<->LDS DMA path (ASYNCcnt-tracked, no VGPR round
// trip): global_load_async_to_lds_b128 -> global_store_async_from_lds_b128.
// ---------------------------------------------------------------------------
__global__ __launch_bounds__(256) void copy_past(const float* __restrict__ Kp,
                                                 const float* __restrict__ Vp,
                                                 float* __restrict__ Ko,
                                                 float* __restrict__ Vo) {
  __shared__ __align__(16) float lbuf[256 * 16];  // 16 KB: 4 x b128 per thread
  const int bh = blockIdx.x;  // 0..127
  const int ch = blockIdx.y;  // 0..63  (4096-float chunks of a 262144-float head)
  const float* __restrict__ src = (blockIdx.z == 0) ? Kp : Vp;
  float* __restrict__ dst = (blockIdx.z == 0) ? Ko : Vo;
  const size_t so = (size_t)bh * TP * DD + (size_t)ch * 4096;
  const size_t dofs = (size_t)bh * TT * DD + (size_t)ch * 4096;
  // LDS byte address of this thread's 64B slot (flat addr low 32 bits == LDS offset)
  const unsigned ldsbase = (unsigned)(uintptr_t)(&lbuf[threadIdx.x * 16]);

#pragma unroll
  for (int u = 0; u < 4; ++u) {
    const float* gs = &src[so + (size_t)(threadIdx.x + u * 256) * 4];
    const unsigned ldsa = ldsbase + u * 16;
    asm volatile("global_load_async_to_lds_b128 %0, %1, off"
                 :: "v"(ldsa), "v"(gs) : "memory");
  }
  asm volatile("s_wait_asynccnt 0" ::: "memory");
#pragma unroll
  for (int u = 0; u < 4; ++u) {
    float* gd = &dst[dofs + (size_t)(threadIdx.x + u * 256) * 4];
    const unsigned ldsa = ldsbase + u * 16;
    asm volatile("global_store_async_from_lds_b128 %0, %1, off"
                 :: "v"(gd), "v"(ldsa) : "memory");
  }
  asm volatile("s_wait_asynccnt 0" ::: "memory");
}

// ---------------------------------------------------------------------------
// 1024x2048x2048 GEMM: Y = X*W + bias, bf16 WMMA, f32 in/out.
// mode 0: Y*(1/sqrt(D)) -> Qbf bf16 (B,H,TN,D)
// mode 1/2: Y -> cache append region (B,H,TT,D) rows TP..TT-1 (f32)
// mode 3: Y -> outF row-major (B*TN, C) (f32)
// ---------------------------------------------------------------------------
__global__ __launch_bounds__(256) void proj_gemm(const float* __restrict__ X,
                                                 const float* __restrict__ W,
                                                 const float* __restrict__ bias,
                                                 int mode,
                                                 float* __restrict__ outF,
                                                 bf16* __restrict__ outQ) {
  __shared__ __align__(16) bf16 As[128][40];  // A tile, row-major, padded
  __shared__ __align__(16) bf16 Bs[128][40];  // B tile transposed: Bs[n][k]

  const int tid = threadIdx.x;
  const int lane = tid & 31;
  const int wid = tid >> 5;
  const int wm = wid & 3;   // 32-row band within 128
  const int wn = wid >> 2;  // 64-col band within 128
  const int hl = lane >> 4;
  const int l15 = lane & 15;

  const int row0 = blockIdx.y * 128;
  const int col0 = blockIdx.x * 128;

  v8f acc[2][4];
#pragma unroll
  for (int i = 0; i < 2; ++i)
#pragma unroll
    for (int j = 0; j < 4; ++j) acc[i][j] = zero8();

  for (int kb = 0; kb < CD; kb += 32) {
    // --- issue all 8 staging loads first (overlap in flight) ---
    float4 xa[4], wb[4];
#pragma unroll
    for (int u = 0; u < 4; ++u) {
      const int un = tid + u * 256;
      xa[u] = *(const float4*)&X[(size_t)(row0 + (un >> 3)) * CD + kb + (un & 7) * 4];
    }
#pragma unroll
    for (int u = 0; u < 4; ++u) {
      const int un = tid + u * 256;
      wb[u] = *(const float4*)&W[(size_t)(kb + (un >> 5)) * CD + col0 + (un & 31) * 4];
    }
    if (kb + 32 < CD) {  // single uniform prefetch of the next k-panel
      __builtin_prefetch(&X[(size_t)(row0 + (tid >> 3)) * CD + kb + 32 + (tid & 7) * 4], 0, 1);
      __builtin_prefetch(&W[(size_t)(kb + 32 + (tid >> 5)) * CD + col0 + (tid & 31) * 4], 0, 1);
    }
    // --- convert + store to LDS ---
#pragma unroll
    for (int u = 0; u < 4; ++u) {
      const int un = tid + u * 256;
      const int r = un >> 3, kq = (un & 7) * 4;
      As[r][kq + 0] = (bf16)xa[u].x;
      As[r][kq + 1] = (bf16)xa[u].y;
      As[r][kq + 2] = (bf16)xa[u].z;
      As[r][kq + 3] = (bf16)xa[u].w;
    }
#pragma unroll
    for (int u = 0; u < 4; ++u) {
      const int un = tid + u * 256;
      const int k = un >> 5, nq = (un & 31) * 4;
      Bs[nq + 0][k] = (bf16)wb[u].x;
      Bs[nq + 1][k] = (bf16)wb[u].y;
      Bs[nq + 2][k] = (bf16)wb[u].z;
      Bs[nq + 3][k] = (bf16)wb[u].w;
    }
    __syncthreads();

    // A frag (16x32 bf16): lane half hl -> K runs [8*hl..8*hl+7], [+16..]
    v16bf afrag[2];
#pragma unroll
    for (int i = 0; i < 2; ++i) {
      const int m = wm * 32 + i * 16 + l15;
      const int k0 = hl * 8;
      afrag[i] = cat16(*(const v8bf*)&As[m][k0], *(const v8bf*)&As[m][k0 + 16]);
    }
    // B frag (32x16 bf16): lane half hl -> K runs [8*hl..8*hl+7], [+16..]
    v16bf bfrag[4];
#pragma unroll
    for (int j = 0; j < 4; ++j) {
      const int n = wn * 64 + j * 16 + l15;
      bfrag[j] = cat16(*(const v8bf*)&Bs[n][hl * 8], *(const v8bf*)&Bs[n][hl * 8 + 16]);
    }
#pragma unroll
    for (int i = 0; i < 2; ++i)
#pragma unroll
      for (int j = 0; j < 4; ++j)
        acc[i][j] = wmma_bf16(afrag[i], bfrag[j], acc[i][j]);
    __syncthreads();
  }

  // Epilogue: C-layout element e -> row = e + 8*hl, col = l15 (per 16x16 tile)
#pragma unroll
  for (int i = 0; i < 2; ++i) {
#pragma unroll
    for (int j = 0; j < 4; ++j) {
      const int col = col0 + wn * 64 + j * 16 + l15;
      const float bv = bias[col];
#pragma unroll
      for (int e = 0; e < 8; ++e) {
        const int row = row0 + wm * 32 + i * 16 + e + hl * 8;
        float v = acc[i][j][e] + bv;
        if (mode == 0) {
          v *= QSCALE;
          const int b = row >> 7, t = row & 127;
          const int h = col >> 7, d = col & 127;
          outQ[(((size_t)(b * HH + h) * TN) + t) * DD + d] = (bf16)v;
        } else if (mode == 3) {
          outF[(size_t)row * CD + col] = v;
        } else {  // K/V append into cache
          const int b = row >> 7, t = row & 127;
          const int h = col >> 7, d = col & 127;
          outF[(((size_t)(b * HH + h) * TT) + TP + t) * DD + d] = v;
        }
      }
    }
  }
}

// ---------------------------------------------------------------------------
// Flash attention: one block per (b,h); 8 waves, each owns a 16-query tile.
// K tiles (32 keys x 128 d) staged row-major, V tiles staged transposed.
// ---------------------------------------------------------------------------
__global__ __launch_bounds__(256) void flash_attn(const bf16* __restrict__ Qbf,
                                                  const float* __restrict__ Kg,
                                                  const float* __restrict__ Vg,
                                                  float* __restrict__ Og) {
  __shared__ __align__(16) bf16 Kt[32][136];     // [key][d], padded
  __shared__ __align__(16) bf16 Vt[128][40];     // transposed [d][key]
  __shared__ __align__(16) bf16 Ps[8][16][40];   // per-wave P relayout buffer

  const int tid = threadIdx.x;
  const int lane = tid & 31;
  const int wid = tid >> 5;
  const int hl = lane >> 4;
  const int l15 = lane & 15;
  const int bh = blockIdx.x;  // b*H + h
  const int qbase = wid * 16;

  // Preload Q as four 16x32 A-fragments (Q already scaled by 1/sqrt(D))
  v16bf qf[4];
  {
    const bf16* __restrict__ qrow = Qbf + ((size_t)bh * TN + qbase + l15) * DD;
#pragma unroll
    for (int kk = 0; kk < 4; ++kk) {
      const int d0 = kk * 32 + hl * 8;
      qf[kk] = cat16(*(const v8bf*)&qrow[d0], *(const v8bf*)&qrow[d0 + 16]);
    }
  }

  v8f oacc[8];
#pragma unroll
  for (int j = 0; j < 8; ++j) oacc[j] = zero8();
  float mrun[8], lrun[8];
#pragma unroll
  for (int e = 0; e < 8; ++e) { mrun[e] = -3.0e38f; lrun[e] = 0.0f; }

  for (int kt = 0; kt < TT / 32; ++kt) {
    const int keyb = kt * 32;
    // --- issue all staging loads first, then convert/store ---
    float4 kv[4], vv[4];
#pragma unroll
    for (int u = 0; u < 4; ++u) {
      const int un = tid + u * 256;
      const size_t base = ((size_t)bh * TT + keyb + (un >> 5)) * DD + (un & 31) * 4;
      kv[u] = *(const float4*)&Kg[base];
      vv[u] = *(const float4*)&Vg[base];
    }
    if (kt + 1 < TT / 32) {
      const size_t nb = ((size_t)bh * TT + keyb + 32 + (tid >> 5)) * DD + (tid & 31) * 4;
      __builtin_prefetch(&Kg[nb], 0, 1);
      __builtin_prefetch(&Vg[nb], 0, 1);
    }
#pragma unroll
    for (int u = 0; u < 4; ++u) {
      const int un = tid + u * 256;
      const int key = un >> 5, dq = (un & 31) * 4;
      Kt[key][dq + 0] = (bf16)kv[u].x;
      Kt[key][dq + 1] = (bf16)kv[u].y;
      Kt[key][dq + 2] = (bf16)kv[u].z;
      Kt[key][dq + 3] = (bf16)kv[u].w;
      Vt[dq + 0][key] = (bf16)vv[u].x;
      Vt[dq + 1][key] = (bf16)vv[u].y;
      Vt[dq + 2][key] = (bf16)vv[u].z;
      Vt[dq + 3][key] = (bf16)vv[u].w;
    }
    __syncthreads();

    // S = Q * K^T : two 16x16 accumulators (key halves), 4 k-chunks of D
    v8f sA = zero8(), sB = zero8();
#pragma unroll
    for (int kk = 0; kk < 4; ++kk) {
      const int koff = kk * 32 + hl * 8;
      const v16bf bA = cat16(*(const v8bf*)&Kt[l15][koff], *(const v8bf*)&Kt[l15][koff + 16]);       // keys keyb+0..15
      const v16bf bB = cat16(*(const v8bf*)&Kt[16 + l15][koff], *(const v8bf*)&Kt[16 + l15][koff + 16]);  // keys keyb+16..31
      sA = wmma_bf16(qf[kk], bA, sA);
      sB = wmma_bf16(qf[kk], bB, sB);
    }

    // Causal mask (only the appended-key tiles can be masked)
    if (keyb + 32 > TP) {
#pragma unroll
      for (int e = 0; e < 8; ++e) {
        const int lim = TP + qbase + e + hl * 8;  // query global position
        if (keyb + l15 > lim) sA[e] = -3.0e38f;
        if (keyb + 16 + l15 > lim) sB[e] = -3.0e38f;
      }
    }

    // Online softmax: row reductions across the 16-lane half (xor 1,2,4,8)
    float cf[8];
#pragma unroll
    for (int e = 0; e < 8; ++e) {
      float mt = fmaxf(sA[e], sB[e]);
      mt = fmaxf(mt, __shfl_xor(mt, 1, 32));
      mt = fmaxf(mt, __shfl_xor(mt, 2, 32));
      mt = fmaxf(mt, __shfl_xor(mt, 4, 32));
      mt = fmaxf(mt, __shfl_xor(mt, 8, 32));
      const float mn = fmaxf(mrun[e], mt);
      cf[e] = __expf(mrun[e] - mn);
      mrun[e] = mn;
      sA[e] = __expf(sA[e] - mn);
      sB[e] = __expf(sB[e] - mn);
      float rs = sA[e] + sB[e];
      rs += __shfl_xor(rs, 1, 32);
      rs += __shfl_xor(rs, 2, 32);
      rs += __shfl_xor(rs, 4, 32);
      rs += __shfl_xor(rs, 8, 32);
      lrun[e] = lrun[e] * cf[e] + rs;
    }
#pragma unroll
    for (int j = 0; j < 8; ++j)
#pragma unroll
      for (int e = 0; e < 8; ++e) oacc[j][e] *= cf[e];

    // Relayout P (C-layout) -> A-fragment via per-wave LDS buffer
#pragma unroll
    for (int e = 0; e < 8; ++e) {
      const int m = e + hl * 8;
      Ps[wid][m][l15] = (bf16)sA[e];
      Ps[wid][m][16 + l15] = (bf16)sB[e];
    }
    asm volatile("s_wait_dscnt 0" ::: "memory");  // wave-local LDS RAW fence
    v16bf pf;
    {
      const int k0 = hl * 8;
      pf = cat16(*(const v8bf*)&Ps[wid][l15][k0],
                 *(const v8bf*)&Ps[wid][l15][k0 + 16]);
    }

    // O += P * V  (V as B-frag from transposed tile: contiguous over key)
#pragma unroll
    for (int j = 0; j < 8; ++j) {
      const v16bf bv = cat16(*(const v8bf*)&Vt[j * 16 + l15][hl * 8], *(const v8bf*)&Vt[j * 16 + l15][hl * 8 + 16]);
      oacc[j] = wmma_bf16(pf, bv, oacc[j]);
    }
    __syncthreads();
  }

  // Normalize and scatter into (B,TN,C) f32 workspace
  const int b = bh >> 4, h = bh & 15;
#pragma unroll
  for (int e = 0; e < 8; ++e) {
    const float inv = 1.0f / lrun[e];
    const int t = qbase + e + hl * 8;
    float* __restrict__ orow = Og + ((size_t)b * TN + t) * CD + h * DD;
#pragma unroll
    for (int j = 0; j < 8; ++j) orow[j * 16 + l15] = oacc[j][e] * inv;
  }
}

// ---------------------------------------------------------------------------
extern "C" void kernel_launch(void* const* d_in, const int* in_sizes, int n_in,
                              void* d_out, int out_size, void* d_ws,
                              size_t ws_size, hipStream_t stream) {
  const float* x  = (const float*)d_in[0];
  const float* Kp = (const float*)d_in[1];
  const float* Vp = (const float*)d_in[2];
  const float* Wq = (const float*)d_in[3];
  const float* bq = (const float*)d_in[4];
  const float* Wk = (const float*)d_in[5];
  const float* bk = (const float*)d_in[6];
  const float* Wv = (const float*)d_in[7];
  const float* bv = (const float*)d_in[8];
  const float* Wo = (const float*)d_in[9];
  const float* bo = (const float*)d_in[10];
  // d_in[11] = position_offset (== TP, baked as compile-time constant)

  float* outO = (float*)d_out;                          // (B,TN,C)
  float* Ko = outO + (size_t)BQ * TN * CD;              // (B,H,TT,D)
  float* Vo = Ko + (size_t)BQ * HH * TT * DD;           // (B,H,TT,D)

  bf16* Qbf = (bf16*)d_ws;                              // (B,H,TN,D) bf16, 4 MB
  float* Oattn =
      (float*)((char*)d_ws + (size_t)BQ * HH * TN * DD * sizeof(bf16));  // 8 MB

  const dim3 blk(256);
  // 1) stream past cache into concatenated K/V outputs (HBM-bound hot path)
  copy_past<<<dim3(BQ * HH, 64, 2), blk, 0, stream>>>(Kp, Vp, Ko, Vo);
  // 2) QKV projections
  const dim3 gg(CD / 128, (BQ * TN) / 128);
  proj_gemm<<<gg, blk, 0, stream>>>(x, Wq, bq, 0, nullptr, Qbf);
  proj_gemm<<<gg, blk, 0, stream>>>(x, Wk, bk, 1, Ko, nullptr);
  proj_gemm<<<gg, blk, 0, stream>>>(x, Wv, bv, 2, Vo, nullptr);
  // 3) flash attention over TT keys
  flash_attn<<<dim3(BQ * HH), blk, 0, stream>>>(Qbf, Ko, Vo, Oattn);
  // 4) output projection
  proj_gemm<<<gg, blk, 0, stream>>>(Oattn, Wo, bo, 3, outO, nullptr);
}
